// RSmatching_model_19980187861329
// MI455X (gfx1250) — matrix operand order repaired and checked by reading, not compiled, not run on hardware
//
#include <hip/hip_runtime.h>
#include <hip/hip_bf16.h>

// Problem constants (from reference): B=64, S=512, D=768, L=2
#define BB   64
#define SS   512
#define DD   768
#define TD   2304          // 3*D
#define KTN  24            // D / 32  (K-chunks per WMMA accumulation)
#define NTN  144           // TD / 16 (N-tiles)
#define LROW 800           // padded LDS row stride (bf16 elems), 1600B = 16B-aligned
// packed-B strides (in bf16 elements)
#define KSTRIDE  73728     // NTN*32*16 : one K-chunk of all N-tiles
#define TSTRIDE  512       // 32*16     : one N-tile

typedef __attribute__((ext_vector_type(16))) __bf16 v16bf;
typedef __attribute__((ext_vector_type(8)))  float  v8f;

union ABu { v16bf v; uint4 q[2]; unsigned short s[16]; };
union Cu  { v8f v; float4 f[2]; };

__device__ __forceinline__ unsigned short f2bf(float f) {
  unsigned int u = __float_as_uint(f);
  unsigned int r = (u + 0x7FFFu + ((u >> 16) & 1u)) >> 16;
  return (unsigned short)r;
}
__device__ __forceinline__ float sigm(float x) { return 1.0f / (1.0f + __expf(-x)); }
#if __has_builtin(__builtin_amdgcn_tanhf)
__device__ __forceinline__ float fast_tanh(float x) { return __builtin_amdgcn_tanhf(x); }
#else
__device__ __forceinline__ float fast_tanh(float x) {
  float e = __expf(-2.0f * x); return (1.0f - e) / (1.0f + e);
}
#endif

// ---------------------------------------------------------------------------
// Pack W (row-major [TD][D] f32) into bf16 WMMA-B tile layout, 32B per lane.
// ---------------------------------------------------------------------------
__global__ void pack_w_kernel(const float* __restrict__ W, unsigned short* __restrict__ dst) {
  int tid = blockIdx.x * blockDim.x + threadIdx.x;   // KTN*NTN*32 threads
  if (tid >= KTN * NTN * 32) return;
  int lane = tid & 31;
  int tile = tid >> 5;
  int nt = tile % NTN;
  int kt = tile / NTN;
  int n  = nt * 16 + (lane & 15);
  int kb = kt * 32 + ((lane >> 4) << 3);
  unsigned short* p = dst + ((size_t)tid << 4);
#pragma unroll
  for (int i = 0; i < 16; ++i)
    p[i] = f2bf(W[(size_t)n * DD + kb + (i < 8 ? i : i + 8)]);
}

// ---------------------------------------------------------------------------
// Phase 1: x_proj = seq @ W_ih^T + b_ih, M reordered as (s, batch-tile) and
// stored fragment-packed: xpf[((mt*NTN+nt)*32+lane)*8 + r], mt = s*4 + btile.
// grid = (2048, 6), block = 256 (8 waves); wave handles 3 adjacent N-tiles.
// ---------------------------------------------------------------------------
__global__ void __launch_bounds__(256)
xproj_kernel(const float* __restrict__ seq, const unsigned short* __restrict__ wih,
             const float* __restrict__ b_ih, float* __restrict__ xpf) {
  __shared__ __align__(16) unsigned short sbuf[16 * LROW];
  const int mt    = blockIdx.x;          // = s*4 + btile
  const int s     = mt >> 2;
  const int btile = mt & 3;
  const int tid   = threadIdx.x;

  // stage 16 rows (batch btile*16.., fixed s), f32 -> bf16
  for (int idx = tid; idx < 16 * DD; idx += 256) {
    int m = idx / DD, k = idx - m * DD;
    sbuf[m * LROW + k] = f2bf(seq[((size_t)(btile * 16 + m) * SS + s) * DD + k]);
  }
  __syncthreads();

  const int lane = tid & 31;
  const int wave = tid >> 5;
  const int nlo  = lane & 15;
  const int hi   = lane >> 4;
  const int nt0  = blockIdx.y * 24 + wave * 3;

  const unsigned short* bp = wih + ((size_t)nt0 * 32 + lane) * 16;
  v8f acc[3] = {};
#pragma unroll 1
  for (int kt = 0; kt < KTN; kt += 2) {
    ABu a0, a1;
    const unsigned short* ar = sbuf + nlo * LROW + kt * 32 + hi * 8;
    a0.q[0] = *(const uint4*)(ar);
    a0.q[1] = *(const uint4*)(ar + 16);
    a1.q[0] = *(const uint4*)(ar + 32);
    a1.q[1] = *(const uint4*)(ar + 48);
    const unsigned short* bk = bp + (size_t)kt * KSTRIDE;
#pragma unroll
    for (int i = 0; i < 3; ++i) {
      ABu b0, b1;
      b0.q[0] = *(const uint4*)(bk + i * TSTRIDE);
      b0.q[1] = *(const uint4*)(bk + i * TSTRIDE + 8);
      b1.q[0] = *(const uint4*)(bk + KSTRIDE + i * TSTRIDE);
      b1.q[1] = *(const uint4*)(bk + KSTRIDE + i * TSTRIDE + 8);
      acc[i] = __builtin_amdgcn_wmma_f32_16x16x32_bf16(false, a0.v, false, b0.v,
                                                       (short)0, acc[i], false, false);
      acc[i] = __builtin_amdgcn_wmma_f32_16x16x32_bf16(false, a1.v, false, b1.v,
                                                       (short)0, acc[i], false, false);
    }
  }
#pragma unroll
  for (int i = 0; i < 3; ++i) {
    int nt = nt0 + i;
    float bias = b_ih[nt * 16 + nlo];
    Cu cu;
#pragma unroll
    for (int r = 0; r < 8; ++r) cu.v[r] = acc[i][r] + bias;
    float* p = xpf + ((size_t)mt * NTN + nt) * 256 + lane * 8;
    *(float4*)(p)     = cu.f[0];
    *(float4*)(p + 4) = cu.f[1];
  }
}

// ---------------------------------------------------------------------------
// Phase 2: masked GRU scan. 4 WGs x 512 threads; WG owns 16 samples.
// Wave w -> N-tiles {w+16j} = 3 (r,z,n) triples; h fp32 in registers, bf16
// copy in LDS as shared WMMA-A. W_hh tiles stream from L2 via one base ptr +
// immediate offsets; gate inputs are fragment-packed b128 loads.
// ---------------------------------------------------------------------------
__global__ void __launch_bounds__(512)
gru_scan_kernel(const float* __restrict__ xpf, const unsigned short* __restrict__ whh,
                const float* __restrict__ b_hh, const int* __restrict__ sot,
                float* __restrict__ hfin) {
  __shared__ __align__(16) unsigned short hbuf[16 * LROW];
  const int tid  = threadIdx.x;
  const int lane = tid & 31;
  const int wave = tid >> 5;
  const int bt   = blockIdx.x;
  const int b0   = bt * 16;
  const int nlo  = lane & 15;
  const int hi   = lane >> 4;

  for (int idx = tid; idx < 16 * LROW; idx += 512) hbuf[idx] = 0;

  float bhr[3], bhz[3], bhn[3];
  int dcol[3];
#pragma unroll
  for (int c = 0; c < 3; ++c) {
    int d = (wave + 16 * c) * 16 + nlo;
    dcol[c] = d;
    bhr[c] = b_hh[d];
    bhz[c] = b_hh[DD + d];
    bhn[c] = b_hh[2 * DD + d];
  }
  const unsigned short* bp = whh + ((size_t)wave * 32 + lane) * 16;
  v8f hreg[3] = {};
  __syncthreads();

  for (int t = 0; t < SS; ++t) {
    // per-step fragment-packed gate-input base (tile mt = t*4+bt, nt0 = wave)
    const float* xbase = xpf + (((size_t)(t * 4 + bt) * NTN + wave) * 32 + lane) * 8;
    __builtin_prefetch(xbase, 0, 1);
    __builtin_prefetch(xbase + 48 * 256, 0, 1);
    __builtin_prefetch(xbase + 96 * 256, 0, 1);

    v8f acc[9] = {};
#pragma unroll 1
    for (int kt = 0; kt < KTN; kt += 2) {
      ABu a0, a1;
      const unsigned short* ar = hbuf + nlo * LROW + kt * 32 + hi * 8;
      a0.q[0] = *(const uint4*)(ar);
      a0.q[1] = *(const uint4*)(ar + 16);
      a1.q[0] = *(const uint4*)(ar + 32);
      a1.q[1] = *(const uint4*)(ar + 48);
      const unsigned short* bk = bp + (size_t)kt * KSTRIDE;
#pragma unroll
      for (int j = 0; j < 9; ++j) {
        ABu b0, b1;
        b0.q[0] = *(const uint4*)(bk + j * 16 * TSTRIDE);
        b0.q[1] = *(const uint4*)(bk + j * 16 * TSTRIDE + 8);
        b1.q[0] = *(const uint4*)(bk + KSTRIDE + j * 16 * TSTRIDE);
        b1.q[1] = *(const uint4*)(bk + KSTRIDE + j * 16 * TSTRIDE + 8);
        acc[j] = __builtin_amdgcn_wmma_f32_16x16x32_bf16(false, a0.v, false, b0.v,
                                                         (short)0, acc[j], false, false);
        acc[j] = __builtin_amdgcn_wmma_f32_16x16x32_bf16(false, a1.v, false, b1.v,
                                                         (short)0, acc[j], false, false);
      }
    }
    // masks for the 8 samples this lane touches (rows r+8*hi), reused by all c
    int mk[8];
#pragma unroll
    for (int r = 0; r < 8; ++r)
      mk[r] = sot[(size_t)(b0 + r + hi * 8) * SS + t];

    // gates: triple (acc[c], acc[c+3], acc[c+6]) = (r,z,n)
#pragma unroll
    for (int c = 0; c < 3; ++c) {
      Cu xr, xz, xn;
      const float* pc = xbase + c * 16 * 256;
      xr.f[0] = *(const float4*)(pc);
      xr.f[1] = *(const float4*)(pc + 4);
      xz.f[0] = *(const float4*)(pc + 48 * 256);
      xz.f[1] = *(const float4*)(pc + 48 * 256 + 4);
      xn.f[0] = *(const float4*)(pc + 96 * 256);
      xn.f[1] = *(const float4*)(pc + 96 * 256 + 4);
#pragma unroll
      for (int r = 0; r < 8; ++r) {
        float rr = sigm(xr.v[r] + acc[c][r]     + bhr[c]);
        float zz = sigm(xz.v[r] + acc[c + 3][r] + bhz[c]);
        float nn = fast_tanh(xn.v[r] + rr * (acc[c + 6][r] + bhn[c]));
        float hnew = (1.0f - zz) * nn + zz * hreg[c][r];
        hreg[c][r] = (mk[r] != 0) ? hnew : hreg[c][r];
      }
    }
    __syncthreads();   // all waves done reading old hbuf
#pragma unroll
    for (int c = 0; c < 3; ++c) {
      int d = dcol[c];
#pragma unroll
      for (int r = 0; r < 8; ++r)
        hbuf[(r + hi * 8) * LROW + d] = f2bf(hreg[c][r]);
    }
    __syncthreads();   // new hbuf visible before next step's WMMA reads
  }

#pragma unroll
  for (int c = 0; c < 3; ++c) {
    int d = dcol[c];
#pragma unroll
    for (int r = 0; r < 8; ++r)
      hfin[(size_t)(b0 + r + hi * 8) * DD + d] = hreg[c][r];
  }
}

// ---------------------------------------------------------------------------
// Phase 3: 2-class head + softmax (negligible)
// ---------------------------------------------------------------------------
__global__ void cls_kernel(const float* __restrict__ hfin, const float* __restrict__ Wc,
                           const float* __restrict__ bc, float* __restrict__ out) {
  int b = threadIdx.x;
  if (b >= BB) return;
  float a0 = bc[0], a1 = bc[1];
  const float* h = hfin + (size_t)b * DD;
  for (int d = 0; d < DD; ++d) {
    float hv = h[d];
    a0 += hv * Wc[d];
    a1 += hv * Wc[DD + d];
  }
  float mx = fmaxf(a0, a1);
  float e0 = __expf(a0 - mx), e1 = __expf(a1 - mx);
  float inv = 1.0f / (e0 + e1);
  out[b * 2 + 0] = e0 * inv;
  out[b * 2 + 1] = e1 * inv;
}

// ---------------------------------------------------------------------------
extern "C" void kernel_launch(void* const* d_in, const int* in_sizes, int n_in,
                              void* d_out, int out_size, void* d_ws, size_t ws_size,
                              hipStream_t stream) {
  (void)in_sizes; (void)n_in; (void)out_size; (void)ws_size;
  const float* seq   = (const float*)d_in[0];
  const int*   sot   = (const int*)  d_in[1];
  const float* W_ih  = (const float*)d_in[2];
  const float* W_hh  = (const float*)d_in[3];
  const float* b_ih  = (const float*)d_in[4];
  const float* b_hh  = (const float*)d_in[5];
  const float* W_cls = (const float*)d_in[6];
  const float* b_cls = (const float*)d_in[7];

  // workspace: [wih packed bf16][whh packed bf16][x_proj fragments f32][h_final f32]
  char* ws = (char*)d_ws;
  const size_t packElems = (size_t)KTN * NTN * 32 * 16;
  unsigned short* wih_p = (unsigned short*)ws;
  unsigned short* whh_p = (unsigned short*)(ws + packElems * 2);
  float* xpf  = (float*)(ws + packElems * 4);
  float* hfin = (float*)(ws + packElems * 4 + (size_t)BB * SS * TD * 4);
  float* out  = (float*)d_out;

  const int packThreads = KTN * NTN * 32;
  pack_w_kernel<<<(packThreads + 255) / 256, 256, 0, stream>>>(W_ih, wih_p);
  pack_w_kernel<<<(packThreads + 255) / 256, 256, 0, stream>>>(W_hh, whh_p);
  xproj_kernel<<<dim3(BB * SS / 16, NTN / 24), 256, 0, stream>>>(seq, wih_p, b_ih, xpf);
  gru_scan_kernel<<<dim3(BB / 16), 512, 0, stream>>>(xpf, whh_p, b_hh, sot, hfin);
  cls_kernel<<<1, 64, 0, stream>>>(hfin, W_cls, b_cls, out);
}
